// TorchNetPoissonProcess_66786741453120
// MI455X (gfx1250) — matrix operand order repaired and checked by reading, not compiled, not run on hardware
//
#include <hip/hip_runtime.h>
#include <math.h>

// ---------------- CDNA5 WMMA types ----------------
typedef __bf16 bf16_t;
typedef bf16_t v16bf __attribute__((ext_vector_type(16)));
typedef float  v8f   __attribute__((ext_vector_type(8)));
typedef unsigned int v4u __attribute__((ext_vector_type(4)));
typedef unsigned int v8u __attribute__((ext_vector_type(8)));

// ---------------- problem constants ----------------
#define NK        64                 // K (features)
#define DD        4                  // window depth
#define NOBS      262144             // N rows of obs
#define TTOT      (NOBS - DD)        // 262140 output timesteps

// ---------------- tiling ----------------
#define THREADS      256
#define ROWS_PER_WG  128             // 8 waves * 16 t-rows
#define NBLOCKS      ((TTOT + ROWS_PER_WG - 1) / ROWS_PER_WG)   // 2048

// ---------------- LDS layout (bytes) ----------------
// B fragments (64KB, TDM-loaded as one flat tile to LDS offset 0):
//   32 frags (4 nt x 8 kc); regions split by (hi/lo) x (v16bf chunk0/chunk1);
//   per region: frag*512 + lane*16  => conflict-free b128, DS-immediate offsets.
#define BH0_OFF      0
#define BH1_OFF      16384
#define BL0_OFF      32768
#define BL1_OFF      49152
#define BFRAG_BYTES  65536
#define A_STRIDE     72              // bf16 per staged obs row (144B, 16B aligned)
#define A_ROWS       132
#define AHI_OFF      65536
#define ALO_OFF      (AHI_OFF + A_ROWS * A_STRIDE * 2)   // 84544
#define ARAW_OFF     (ALO_OFF + A_ROWS * A_STRIDE * 2)   // 103552 (raw f32 obs tile, TDM)
#define SMEM_BYTES   (ARAW_OFF + A_ROWS * NK * 4)        // 137344

// workspace layout: [0,64K) = pre-swizzled B-fragment block; partials after
#define WS_PART_OFF_FLOATS  (BFRAG_BYTES / 4)            // 16384

union V16 { v16bf v; uint4 q[2]; };
union BF16B { bf16_t b; unsigned short u; };

// ---- Tensor Data Mover: 2D tile (dim0 elems/row) from global -> LDS ----
__device__ __forceinline__ void tdm_load_2d(unsigned lds_off, const void* gptr,
                                            unsigned dim0, unsigned dim1,
                                            unsigned tile0, unsigned tile1,
                                            unsigned stride0)
{
    unsigned long long ga = (unsigned long long)(size_t)gptr;
    v4u g0;
    g0.x = 1u;                                    // count=1 user descriptor
    g0.y = lds_off;                               // lds_addr
    g0.z = (unsigned)ga;                          // global_addr[31:0]
    g0.w = ((unsigned)(ga >> 32) & 0x1FFFFFFu)    // global_addr[56:32]
         | 0x80000000u;                           // type=2 ("image")
    v8u g1;
    g1.s0 = 2u << 16;                             // data_size=2 (4 bytes)
    g1.s1 = (dim0 & 0xFFFFu) << 16;               // tensor_dim0 lo
    g1.s2 = (dim0 >> 16) | ((dim1 & 0xFFFFu) << 16);
    g1.s3 = (dim1 >> 16) | (tile0 << 16);         // tensor_dim1 hi | tile_dim0
    g1.s4 = tile1;                                // tile_dim1 (tile_dim2=0)
    g1.s5 = stride0;                              // tensor_dim0_stride lo32
    g1.s6 = 0u;
    g1.s7 = 0u;
    asm volatile("tensor_load_to_lds %0, %1" :: "s"(g0), "s"(g1) : "memory");
}

// ---- one-shot prep: build pre-swizzled hi/lo bf16 B fragments in workspace ----
// B layout used here (contiguous k, A matches): lane L holds column n=L&15,
// K_local=(L>>4)*16+p; chunk0 = p 0..7, chunk1 = p 8..15.
__global__ void poisson_prep_kernel(const float* __restrict__ beta,
                                    unsigned short* __restrict__ bws)
{
    int f    = blockIdx.x * blockDim.x + threadIdx.x;     // 0..16383
    int nt   = f >> 12;
    int kc   = (f >> 9) & 7;
    int ln   = (f >> 4) & 31;
    int p    = f & 15;
    int Kg   = kc * 32 + (ln >> 4) * 16 + p;              // 0..255 = i*64 + j
    int i    = Kg >> 6;
    int j    = Kg & 63;
    int kcol = nt * 16 + (ln & 15);
    float v  = beta[NK + (kcol * NK + j) * DD + i];       // B3[kcol, j, i]
    BF16B hi, lo;
    hi.b = (bf16_t)v;
    lo.b = (bf16_t)(v - (float)hi.b);
    int slot = ((nt * 8 + kc) * 32 + ln) * 8 + (p & 7);
    if (p < 8) { bws[slot]         = hi.u; bws[16384 + slot] = lo.u; }
    else       { bws[8192 + slot]  = hi.u; bws[24576 + slot] = lo.u; }
}

__launch_bounds__(THREADS)
__global__ void poisson_lam_kernel(const float* __restrict__ obs,
                                   const float* __restrict__ beta,
                                   const void* __restrict__ bfrag,   // ws bytes [0,64K)
                                   float* __restrict__ out,          // [0]=loglik, [1..]=lam
                                   float* __restrict__ part_ws)      // per-block partials
{
    __shared__ __align__(16) unsigned char smem[SMEM_BYTES];
    __shared__ float red[THREADS];

    const int tid  = threadIdx.x;
    const int t0   = blockIdx.x * ROWS_PER_WG;
    const int wave = tid >> 5;
    const int lane = tid & 31;

    // ---- TDM: B-fragment block (flat 64KB) + obs window tile (132 x 64 f32) ----
    if (wave == 0) {
        tdm_load_2d((unsigned)(size_t)(smem + BH0_OFF), bfrag,
                    BFRAG_BYTES / 4, 1, BFRAG_BYTES / 4, 1, BFRAG_BYTES / 4);
        tdm_load_2d((unsigned)(size_t)(smem + ARAW_OFF), obs + (size_t)t0 * NK,
                    NK, NOBS, NK, A_ROWS, NK);      // OOB rows read as zero (tail)
        __builtin_amdgcn_s_wait_tensorcnt(0);
    }
    __syncthreads();

    // ---- convert the TDM-staged f32 tile to bf16 hi/lo (fragment-friendly rows) ----
    const float* Araw = (const float*)(smem + ARAW_OFF);
    bf16_t* Ahi = (bf16_t*)(smem + AHI_OFF);
    bf16_t* Alo = (bf16_t*)(smem + ALO_OFF);
    for (int e = tid; e < A_ROWS * NK; e += THREADS) {
        int row = e >> 6;
        int col = e & 63;
        float x   = Araw[e];
        bf16_t hi = (bf16_t)x;
        bf16_t lo = (bf16_t)(x - (float)hi);
        Ahi[row * A_STRIDE + col] = hi;
        Alo[row * A_STRIDE + col] = lo;
    }
    __syncthreads();

    const int m    = lane & 15;
    const int half = lane >> 4;

    v8f acc[4];
    #pragma unroll
    for (int nt = 0; nt < 4; ++nt)
        #pragma unroll
        for (int q = 0; q < 8; ++q) acc[nt][q] = 0.f;

    // hoisted per-lane LDS base pointers (hot-loop offsets are DS immediates)
    const unsigned char* pA   = smem + AHI_OFF + ((wave * 16 + m) * A_STRIDE + half * 16) * 2;
    const unsigned char* pBh0 = smem + BH0_OFF + lane * 16;
    const unsigned char* pBh1 = smem + BH1_OFF + lane * 16;
    const unsigned char* pBl0 = smem + BL0_OFF + lane * 16;
    const unsigned char* pBl1 = smem + BL1_OFF + lane * 16;

    // ---- K=256 contraction: 8 steps of K=32; bf16 split => hi*hi + hi*lo + lo*hi ----
    #pragma unroll
    for (int kc = 0; kc < 8; ++kc) {
        const int aoff = (kc >> 1) * (A_STRIDE * 2) + (kc & 1) * 64;   // const after unroll
        V16 a_hi, a_lo;
        a_hi.q[0] = *(const uint4*)(pA + aoff);
        a_hi.q[1] = *(const uint4*)(pA + aoff + 16);
        a_lo.q[0] = *(const uint4*)(pA + aoff + (ALO_OFF - AHI_OFF));
        a_lo.q[1] = *(const uint4*)(pA + aoff + (ALO_OFF - AHI_OFF) + 16);

        V16 bh[4], bl[4];
        #pragma unroll
        for (int nt = 0; nt < 4; ++nt) {
            const int boff = (nt * 8 + kc) * 512;                      // <= 15872: DS imm
            bh[nt].q[0] = *(const uint4*)(pBh0 + boff);
            bh[nt].q[1] = *(const uint4*)(pBh1 + boff);
            bl[nt].q[0] = *(const uint4*)(pBl0 + boff);
            bl[nt].q[1] = *(const uint4*)(pBl1 + boff);
        }
        // round-robin across 4 independent accumulators: same-acc dep distance = 4
        #pragma unroll
        for (int nt = 0; nt < 4; ++nt)
            acc[nt] = __builtin_amdgcn_wmma_f32_16x16x32_bf16(
                false, a_hi.v, false, bh[nt].v, (short)0, acc[nt], false, false);
        #pragma unroll
        for (int nt = 0; nt < 4; ++nt)
            acc[nt] = __builtin_amdgcn_wmma_f32_16x16x32_bf16(
                false, a_hi.v, false, bl[nt].v, (short)0, acc[nt], false, false);
        #pragma unroll
        for (int nt = 0; nt < 4; ++nt)
            acc[nt] = __builtin_amdgcn_wmma_f32_16x16x32_bf16(
                false, a_lo.v, false, bh[nt].v, (short)0, acc[nt], false, false);
    }

    // ---- epilogue: lam = acc + beta0; store; loglik terms (tgt from LDS tile) ----
    // C/D layout: VGPR r, lane L -> row M = (L>>4)*8 + r, col N = L&15
    float part = 0.f;
    #pragma unroll
    for (int nt = 0; nt < 4; ++nt) {
        int k    = nt * 16 + (lane & 15);
        float b0 = beta[k];
        #pragma unroll
        for (int r = 0; r < 8; ++r) {
            int tl = wave * 16 + half * 8 + r;     // local t-row
            int t  = t0 + tl;
            float lam = acc[nt][r] + b0;
            if (t < TTOT) {
                out[1 + t * NK + k] = lam;
                float tgt = Araw[(tl + DD) * NK + k];   // exact f32 from TDM tile
                part += tgt * logf(lam) - lam;
            }
        }
    }

    red[tid] = part;
    __syncthreads();
    #pragma unroll
    for (int s = THREADS / 2; s > 0; s >>= 1) {
        if (tid < s) red[tid] += red[tid + s];
        __syncthreads();
    }
    if (tid == 0) part_ws[blockIdx.x] = red[0];
}

// Deterministic final reduction (double accumulation), writes out[0] each launch.
__global__ void poisson_reduce_kernel(const float* __restrict__ part_ws,
                                      float* __restrict__ out, int n)
{
    __shared__ double sd[256];
    double s = 0.0;
    for (int i = threadIdx.x; i < n; i += 256) s += (double)part_ws[i];
    sd[threadIdx.x] = s;
    __syncthreads();
    for (int k = 128; k > 0; k >>= 1) {
        if ((int)threadIdx.x < k) sd[threadIdx.x] += sd[threadIdx.x + k];
        __syncthreads();
    }
    if (threadIdx.x == 0) out[0] = (float)sd[0];
}

extern "C" void kernel_launch(void* const* d_in, const int* in_sizes, int n_in,
                              void* d_out, int out_size, void* d_ws, size_t ws_size,
                              hipStream_t stream)
{
    const float* obs  = (const float*)d_in[0];   // (N, 64) f32
    const float* beta = (const float*)d_in[1];   // (1, 16448) f32
    float* out = (float*)d_out;                  // [0]=loglik, [1..]=lam (T*64)
    unsigned short* bws = (unsigned short*)d_ws; // [0,64K): swizzled B fragments
    float* part = (float*)d_ws + WS_PART_OFF_FLOATS;   // NBLOCKS partials (8 KB)

    poisson_prep_kernel<<<16384 / THREADS, THREADS, 0, stream>>>(beta, bws);
    poisson_lam_kernel<<<NBLOCKS, THREADS, 0, stream>>>(obs, beta, d_ws, out, part);
    poisson_reduce_kernel<<<1, 256, 0, stream>>>(part, out, NBLOCKS);
}
